// GATEncoder_43911745634520
// MI455X (gfx1250) — matrix operand order repaired and checked by reading, not compiled, not run on hardware
//
#include <hip/hip_runtime.h>
#include <cmath>
#include <cstdint>

// ---------------------------------------------------------------------------
// Types for CDNA5 WMMA (wave32)
// ---------------------------------------------------------------------------
typedef __attribute__((ext_vector_type(16))) __bf16         v16bf;
typedef __attribute__((ext_vector_type(16))) unsigned short v16us;
typedef __attribute__((ext_vector_type(8)))  float          v8f;

__device__ __forceinline__ unsigned short f2bf(float f) {
  // round-to-nearest-even fp32 -> bf16 (weight conversion path)
  unsigned int u = __float_as_uint(f);
  u += 0x7FFFu + ((u >> 16) & 1u);
  return (unsigned short)(u >> 16);
}

// native fp32 -> bf16 conversion (lowers to v_cvt_pk_bf16_f32)
__device__ __forceinline__ v16bf conv16(const float4 x0, const float4 x1,
                                        const float4 x2, const float4 x3) {
  v16bf r;
  r[0]  = (__bf16)x0.x; r[1]  = (__bf16)x0.y; r[2]  = (__bf16)x0.z; r[3]  = (__bf16)x0.w;
  r[4]  = (__bf16)x1.x; r[5]  = (__bf16)x1.y; r[6]  = (__bf16)x1.z; r[7]  = (__bf16)x1.w;
  r[8]  = (__bf16)x2.x; r[9]  = (__bf16)x2.y; r[10] = (__bf16)x2.z; r[11] = (__bf16)x2.w;
  r[12] = (__bf16)x3.x; r[13] = (__bf16)x3.y; r[14] = (__bf16)x3.z; r[15] = (__bf16)x3.w;
  return r;
}

__device__ __forceinline__ float elu_f(float x) { return x > 0.f ? x : (expf(x) - 1.f); }

__device__ __forceinline__ void atomicMaxF32(float* addr, float val) {
  if (val >= 0.f) atomicMax((int*)addr, __float_as_int(val));
  else            atomicMin((unsigned int*)addr, __float_as_uint(val));
}

__device__ __forceinline__ void load_a(const float* __restrict__ p, int koff,
                                       int ka, int kb, float4 out[4]) {
  out[0] = *(const float4*)(p + koff + ka);
  out[1] = *(const float4*)(p + koff + ka + 4);
  out[2] = *(const float4*)(p + koff + kb);
  out[3] = *(const float4*)(p + koff + kb + 4);
}
__device__ __forceinline__ void load_b(const unsigned short* __restrict__ Bp,
                                       int K, int koff, v16us out[4]) {
#pragma unroll
  for (int s = 0; s < 4; ++s) out[s] = *(const v16us*)(Bp + (size_t)(s * 16) * K + koff);
}

// ---------------------------------------------------------------------------
// WMMA GEMM: C[M,Nc] = A[M,K] (f32, row-major, lda==K) x B[K,Nc]
// B pre-transposed + bf16-converted: Bt[Nc][K] (ushort-as-bf16).
// One wave computes a 32x64 tile. k-loop unrolled by 64 with ping-pong
// A/B staging buffers: 16 v_wmma_f32_16x16x32_bf16 per unrolled body.
// Requires K % 64 == 0 (true here: 256 / 512 / 640).
// waves_per_eu(1): allow the allocator to trade occupancy for registers —
// the WMMA pipe hides its own latency; spilling accumulators would not.
// ---------------------------------------------------------------------------
__global__ __launch_bounds__(256)
__attribute__((amdgpu_waves_per_eu(1)))
void k_gemm_bf16(const float* __restrict__ A,
                 const unsigned short* __restrict__ Bt,
                 float* __restrict__ C, int M, int Nc, int K) {
  const int lane = threadIdx.x & 31;
  const int wid  = blockIdx.x * (blockDim.x >> 5) + (threadIdx.x >> 5);
  const int tiles_n = Nc >> 6;          // 64-wide wave tiles
  const int tiles_m = (M + 31) >> 5;    // 32-high wave tiles
  if (wid >= tiles_m * tiles_n) return;
  const int mt = wid / tiles_n, nt = wid % tiles_n;
  const int hf  = lane >> 4;            // wave half
  const int l15 = lane & 15;
  const int m0  = mt * 32;

  int r0 = m0 + l15;      if (r0 >= M) r0 = M - 1;
  int r1 = m0 + 16 + l15; if (r1 >= M) r1 = M - 1;
  const float* Ap0 = A + (size_t)r0 * K;
  const float* Ap1 = A + (size_t)r1 * K;
  const int ka = hf * 16;               // A-layout (contiguous): elems 0..7  = K ka..ka+7
  const int kb = hf * 16 + 8;           //                        elems 8..15 = K kb..kb+7
  const unsigned short* Bp = Bt + (size_t)(nt * 64 + l15) * K + hf * 16;

  const v8f zero = {0.f,0.f,0.f,0.f,0.f,0.f,0.f,0.f};
  v8f acc[8] = {zero, zero, zero, zero, zero, zero, zero, zero};

  // ping-pong staging buffers
  float4 aA0[4], aA1[4], aB0[4], aB1[4];
  v16us  bA[4], bB[4];
  load_a(Ap0, 0, ka, kb, aA0);
  load_a(Ap1, 0, ka, kb, aA1);
  load_b(Bp, K, 0, bA);

  for (int kk = 0; kk < K; kk += 64) {
    // issue loads for the second half-step, then compute on buffer A
    const int k1 = kk + 32;
    load_a(Ap0, k1, ka, kb, aB0);
    load_a(Ap1, k1, ka, kb, aB1);
    load_b(Bp, K, k1, bB);
    {
      const v16bf af0 = conv16(aA0[0], aA0[1], aA0[2], aA0[3]);
      const v16bf af1 = conv16(aA1[0], aA1[1], aA1[2], aA1[3]);
#pragma unroll
      for (int s = 0; s < 4; ++s) {
        const v16bf bf = __builtin_bit_cast(v16bf, bA[s]);
        acc[s]     = __builtin_amdgcn_wmma_f32_16x16x32_bf16(
            false, af0, false, bf, (short)0, acc[s],     false, false);
        acc[4 + s] = __builtin_amdgcn_wmma_f32_16x16x32_bf16(
            false, af1, false, bf, (short)0, acc[4 + s], false, false);
      }
    }
    // issue loads for the next iteration's first half, compute on buffer B
    const int k2 = kk + 64;
    if (k2 < K) {
      load_a(Ap0, k2, ka, kb, aA0);
      load_a(Ap1, k2, ka, kb, aA1);
      load_b(Bp, K, k2, bA);
    }
    {
      const v16bf af0 = conv16(aB0[0], aB0[1], aB0[2], aB0[3]);
      const v16bf af1 = conv16(aB1[0], aB1[1], aB1[2], aB1[3]);
#pragma unroll
      for (int s = 0; s < 4; ++s) {
        const v16bf bf = __builtin_bit_cast(v16bf, bB[s]);
        acc[s]     = __builtin_amdgcn_wmma_f32_16x16x32_bf16(
            false, af0, false, bf, (short)0, acc[s],     false, false);
        acc[4 + s] = __builtin_amdgcn_wmma_f32_16x16x32_bf16(
            false, af1, false, bf, (short)0, acc[4 + s], false, false);
      }
    }
  }

  // --- store: C layout: vgpr r -> row r (+8 for hi lane half) ---
  const int colb = nt * 64 + l15;
  const int rb0  = m0 + (hf ? 8 : 0);
  const int rb1  = m0 + 16 + (hf ? 8 : 0);
#pragma unroll
  for (int s = 0; s < 4; ++s) {
#pragma unroll
    for (int r = 0; r < 8; ++r) {
      const int row0 = rb0 + r;
      const int row1 = rb1 + r;
      if (row0 < M) C[(size_t)row0 * Nc + colb + s * 16] = acc[s][r];
      if (row1 < M) C[(size_t)row1 * Nc + colb + s * 16] = acc[4 + s][r];
    }
  }
}

// ---------------------------------------------------------------------------
// Weight transpose + bf16 convert: W[K,Nc] f32 -> Bt[Nc,K] bf16(ushort)
// ---------------------------------------------------------------------------
__global__ void k_w2bf(const float* __restrict__ W, unsigned short* __restrict__ Bt,
                       int K, int Nc) {
  size_t tid = (size_t)blockIdx.x * blockDim.x + threadIdx.x;
  if (tid >= (size_t)K * Nc) return;
  int k = (int)(tid / Nc), n = (int)(tid % Nc);
  Bt[(size_t)n * K + k] = f2bf(W[tid]);
}

// ---------------------------------------------------------------------------
// LayerNorm: one wave per row (D = 512 -> 16 elems/lane), wave32 shuffles
// ---------------------------------------------------------------------------
__global__ void k_layernorm(const float* __restrict__ x, float* __restrict__ y,
                            const float* __restrict__ g, const float* __restrict__ b,
                            int N, int D) {
  int row = blockIdx.x * (blockDim.x >> 5) + (threadIdx.x >> 5);
  if (row >= N) return;
  int lane = threadIdx.x & 31;
  const float* xr = x + (size_t)row * D;
  float v[16];
  const int per = D >> 5;
  float s = 0.f;
  for (int i = 0; i < per; ++i) { v[i] = xr[lane + i * 32]; s += v[i]; }
  for (int o = 16; o; o >>= 1) s += __shfl_xor(s, o, 32);
  float mu = s / (float)D;
  float q = 0.f;
  for (int i = 0; i < per; ++i) { float t = v[i] - mu; q += t * t; }
  for (int o = 16; o; o >>= 1) q += __shfl_xor(q, o, 32);
  float inv = rsqrtf(q / (float)D + 1e-5f);
  float* yr = y + (size_t)row * D;
  for (int i = 0; i < per; ++i) {
    int c = lane + i * 32;
    yr[c] = (v[i] - mu) * inv * g[c] + b[c];
  }
}

// ---------------------------------------------------------------------------
// Attention helper kernels
// ---------------------------------------------------------------------------
__global__ void k_attn_node(const float* __restrict__ h, const float* __restrict__ a_s,
                            const float* __restrict__ a_d, float* __restrict__ asn,
                            float* __restrict__ adn, int N, int H, int C, int ldh) {
  size_t tid = (size_t)blockIdx.x * blockDim.x + threadIdx.x;
  if (tid >= (size_t)N * H) return;
  int n = (int)(tid / H), hh = (int)(tid % H);
  const float* hp  = h + (size_t)n * ldh + hh * C;
  const float* asp = a_s + hh * C;
  const float* adp = a_d + hh * C;
  float ss = 0.f, sd = 0.f;
  for (int c = 0; c < C; ++c) { float v = hp[c]; ss += v * asp[c]; sd += v * adp[c]; }
  asn[tid] = ss; adn[tid] = sd;
}

__global__ void k_edge_e(const float* __restrict__ asn, const float* __restrict__ adn,
                         const int* __restrict__ src, const int* __restrict__ dst,
                         float* __restrict__ e, float* __restrict__ m,
                         long long EH, int H) {
  size_t tid = (size_t)blockIdx.x * blockDim.x + threadIdx.x;
  if (tid >= (size_t)EH) return;
  int ed = (int)(tid / H), hh = (int)(tid % H);
  int s = src[ed], d = dst[ed];
  float v = asn[(size_t)s * H + hh] + adn[(size_t)d * H + hh];
  v = v > 0.f ? v : 0.2f * v;                 // leaky_relu(0.2)
  e[tid] = v;
  atomicMaxF32(&m[(size_t)d * H + hh], v);
}

__global__ void k_edge_ex(float* __restrict__ e, const float* __restrict__ m,
                          float* __restrict__ den, const int* __restrict__ dst,
                          long long EH, int H) {
  size_t tid = (size_t)blockIdx.x * blockDim.x + threadIdx.x;
  if (tid >= (size_t)EH) return;
  int ed = (int)(tid / H), hh = (int)(tid % H);
  int d = dst[ed];
  float v = expf(e[tid] - m[(size_t)d * H + hh]);
  e[tid] = v;
  atomicAdd(&den[(size_t)d * H + hh], v);
}

// out[dst, col0 + c] += (ex/den) * h[src, c]   (concat heads)
__global__ void k_scatter(const float* __restrict__ h, const float* __restrict__ ex,
                          const float* __restrict__ den, const int* __restrict__ src,
                          const int* __restrict__ dst, float* __restrict__ out,
                          long long total, int H, int C, int ldh, int ldo, int col0) {
  size_t tid = (size_t)blockIdx.x * blockDim.x + threadIdx.x;
  if (tid >= (size_t)total) return;
  const int HC = H * C;
  int ed = (int)(tid / HC), c = (int)(tid % HC);
  int hh = c / C;
  int s = src[ed], d = dst[ed];
  float coeff = ex[(size_t)ed * H + hh] / den[(size_t)d * H + hh];
  atomicAdd(&out[(size_t)d * ldo + col0 + c], coeff * h[(size_t)s * ldh + c]);
}

// mean over heads (final GAT, concat=False)
__global__ void k_scatter_mean(const float* __restrict__ h, const float* __restrict__ ex,
                               const float* __restrict__ den, const int* __restrict__ src,
                               const int* __restrict__ dst, float* __restrict__ out,
                               long long total, int H, int C, int ldh) {
  size_t tid = (size_t)blockIdx.x * blockDim.x + threadIdx.x;
  if (tid >= (size_t)total) return;
  int ed = (int)(tid / C), c = (int)(tid % C);
  int s = src[ed], d = dst[ed];
  float acc = 0.f;
  for (int hh = 0; hh < H; ++hh) {
    float coeff = ex[(size_t)ed * H + hh] / den[(size_t)d * H + hh];
    acc += coeff * h[(size_t)s * ldh + hh * C + c];
  }
  atomicAdd(&out[(size_t)d * C + c], acc / (float)H);
}

// ---------------------------------------------------------------------------
// Elementwise kernels
// ---------------------------------------------------------------------------
__global__ void k_fill(float* p, float v, long long n) {
  size_t tid = (size_t)blockIdx.x * blockDim.x + threadIdx.x;
  if (tid < (size_t)n) p[tid] = v;
}
__global__ void k_bias_elu(const float* __restrict__ in, const float* __restrict__ b,
                           float* __restrict__ out, long long n, int D) {
  size_t tid = (size_t)blockIdx.x * blockDim.x + threadIdx.x;
  if (tid < (size_t)n) out[tid] = elu_f(in[tid] + b[tid % D]);
}
__global__ void k_xc_bias_elu(float* __restrict__ xc, const float* __restrict__ b1,
                              const float* __restrict__ b2, long long n) {
  size_t tid = (size_t)blockIdx.x * blockDim.x + threadIdx.x;
  if (tid >= (size_t)n) return;
  int c = (int)(tid % 640);
  float b = c < 512 ? b1[c] : b2[c - 512];
  xc[tid] = elu_f(xc[tid] + b);
}
__global__ void k_resid_elu(float* __restrict__ xa, const float* __restrict__ t,
                            const float* __restrict__ b, long long n, int D) {
  size_t tid = (size_t)blockIdx.x * blockDim.x + threadIdx.x;
  if (tid < (size_t)n) xa[tid] = elu_f(xa[tid] + t[tid] + b[tid % D]);
}
__global__ void k_out_bias(float* __restrict__ out, const float* __restrict__ b,
                           long long n, int C) {
  size_t tid = (size_t)blockIdx.x * blockDim.x + threadIdx.x;
  if (tid < (size_t)n) out[tid] = b[tid % C];
}
__global__ void k_selfloops(const int* __restrict__ ei, int* __restrict__ s2,
                            int* __restrict__ d2, int E, int E2) {
  size_t tid = (size_t)blockIdx.x * blockDim.x + threadIdx.x;
  if (tid >= (size_t)E2) return;
  if (tid < (size_t)E) { s2[tid] = ei[tid]; d2[tid] = ei[(size_t)E + tid]; }
  else { int n = (int)(tid - E); s2[tid] = n; d2[tid] = n; }
}

// ---------------------------------------------------------------------------
// Host orchestration
// ---------------------------------------------------------------------------
static inline long long cdivll(long long a, long long b) { return (a + b - 1) / b; }

extern "C" void kernel_launch(void* const* d_in, const int* in_sizes, int n_in,
                              void* d_out, int out_size, void* d_ws, size_t ws_size,
                              hipStream_t stream) {
  const float* x_in   = (const float*)d_in[0];
  const int*   ei     = (const int*)d_in[1];
  const float* proj_W = (const float*)d_in[2];
  const float* proj_b = (const float*)d_in[3];
  const float* blk_g  = (const float*)d_in[4];
  const float* blk_b  = (const float*)d_in[5];
  const float* g1_W   = (const float*)d_in[6];
  const float* g1_as  = (const float*)d_in[7];
  const float* g1_ad  = (const float*)d_in[8];
  const float* g1_b   = (const float*)d_in[9];
  const float* g2_W   = (const float*)d_in[10];
  const float* g2_as  = (const float*)d_in[11];
  const float* g2_ad  = (const float*)d_in[12];
  const float* g2_b   = (const float*)d_in[13];
  const float* uni_W  = (const float*)d_in[14];
  const float* uni_b  = (const float*)d_in[15];
  const float* fin_g  = (const float*)d_in[16];
  const float* fin_b  = (const float*)d_in[17];
  const float* fg_W   = (const float*)d_in[18];
  const float* fg_as  = (const float*)d_in[19];
  const float* fg_ad  = (const float*)d_in[20];
  const float* fg_b   = (const float*)d_in[21];
  float* out = (float*)d_out;

  const int N  = in_sizes[0] / 256;
  const int E  = in_sizes[1] / 2;
  const int E2 = E + N;           // + self loops

  // ---- carve workspace ----
  char* ws = (char*)d_ws;
  size_t off = 0;
  auto carve = [&](size_t bytes) -> char* {
    char* p = ws + off; off = (off + bytes + 255) & ~(size_t)255; return p;
  };
  unsigned short* projWt = (unsigned short*)carve((size_t)256 * 512 * 2);
  unsigned short* g1Wt   = (unsigned short*)carve((size_t)3 * 512 * 512 * 2);
  unsigned short* g2Wt   = (unsigned short*)carve((size_t)3 * 512 * 128 * 2);
  unsigned short* uniWt  = (unsigned short*)carve((size_t)3 * 640 * 512 * 2);
  unsigned short* fgWt   = (unsigned short*)carve((size_t)512 * 1024 * 2);
  float* xa   = (float*)carve((size_t)N * 512 * 4);
  float* xn   = (float*)carve((size_t)N * 512 * 4);
  float* xc   = (float*)carve((size_t)N * 640 * 4);
  float* hblk = (float*)carve((size_t)N * 1024 * 4);   // h1 = hblk, h2 after it
  float* h1 = hblk;
  float* h2 = hblk + (size_t)N * 512;
  float* asn = (float*)carve((size_t)N * 8 * 4);
  float* adn = (float*)carve((size_t)N * 8 * 4);
  float* mmx = (float*)carve((size_t)N * 8 * 4);
  float* den = (float*)carve((size_t)N * 8 * 4);
  float* eE  = (float*)carve((size_t)E2 * 8 * 4);
  int* src2  = (int*)carve((size_t)E2 * 4);
  int* dst2  = (int*)carve((size_t)E2 * 4);

  const int B = 256;
  auto g1d = [&](long long n) { return dim3((unsigned)cdivll(n, B)); };
  auto gemm = [&](const float* A, const unsigned short* Bt, float* C,
                  int M, int Nc, int K) {
    long long waves = cdivll(M, 32) * (Nc / 64);
    k_gemm_bf16<<<dim3((unsigned)cdivll(waves, 8)), B, 0, stream>>>(A, Bt, C, M, Nc, K);
  };

  // ---- prep: self loops + bf16 transposed weights ----
  k_selfloops<<<g1d(E2), B, 0, stream>>>(ei, src2, dst2, E, E2);
  k_w2bf<<<g1d((long long)256 * 512), B, 0, stream>>>(proj_W, projWt, 256, 512);
  for (int i = 0; i < 3; ++i) {
    k_w2bf<<<g1d((long long)512 * 512), B, 0, stream>>>(g1_W + (size_t)i * 512 * 512,
                                                        g1Wt + (size_t)i * 512 * 512, 512, 512);
    k_w2bf<<<g1d((long long)512 * 128), B, 0, stream>>>(g2_W + (size_t)i * 512 * 128,
                                                        g2Wt + (size_t)i * 512 * 128, 512, 128);
    k_w2bf<<<g1d((long long)640 * 512), B, 0, stream>>>(uni_W + (size_t)i * 640 * 512,
                                                        uniWt + (size_t)i * 640 * 512, 640, 512);
  }
  k_w2bf<<<g1d((long long)512 * 1024), B, 0, stream>>>(fg_W, fgWt, 512, 1024);

  // ---- projection: xa = elu(x @ proj_W + proj_b) ----
  gemm(x_in, projWt, h1, N, 512, 256);
  k_bias_elu<<<g1d((long long)N * 512), B, 0, stream>>>(h1, proj_b, xa, (long long)N * 512, 512);

  // ---- 3 GAT blocks ----
  for (int i = 0; i < 3; ++i) {
    k_layernorm<<<g1d((long long)N * 32), B, 0, stream>>>(xa, xn, blk_g + (size_t)i * 512,
                                                          blk_b + (size_t)i * 512, N, 512);
    gemm(xn, g1Wt + (size_t)i * 512 * 512, h1, N, 512, 512);
    gemm(xn, g2Wt + (size_t)i * 512 * 128, h2, N, 128, 512);
    k_fill<<<g1d((long long)N * 640), B, 0, stream>>>(xc, 0.f, (long long)N * 640);

    // GAT1: H=4, C=128, ldh=512 -> xc[:, 0:512]
    {
      const int H = 4, C = 128;
      k_attn_node<<<g1d((long long)N * H), B, 0, stream>>>(h1, g1_as + (size_t)i * 512,
                                                           g1_ad + (size_t)i * 512,
                                                           asn, adn, N, H, C, 512);
      k_fill<<<g1d((long long)N * H), B, 0, stream>>>(mmx, -INFINITY, (long long)N * H);
      k_fill<<<g1d((long long)N * H), B, 0, stream>>>(den, 0.f, (long long)N * H);
      k_edge_e<<<g1d((long long)E2 * H), B, 0, stream>>>(asn, adn, src2, dst2, eE, mmx,
                                                         (long long)E2 * H, H);
      k_edge_ex<<<g1d((long long)E2 * H), B, 0, stream>>>(eE, mmx, den, dst2,
                                                          (long long)E2 * H, H);
      k_scatter<<<g1d((long long)E2 * H * C), B, 0, stream>>>(h1, eE, den, src2, dst2, xc,
                                                              (long long)E2 * H * C, H, C,
                                                              512, 640, 0);
    }
    // GAT2: H=2, C=64, ldh=128 -> xc[:, 512:640]
    {
      const int H = 2, C = 64;
      k_attn_node<<<g1d((long long)N * H), B, 0, stream>>>(h2, g2_as + (size_t)i * 128,
                                                           g2_ad + (size_t)i * 128,
                                                           asn, adn, N, H, C, 128);
      k_fill<<<g1d((long long)N * H), B, 0, stream>>>(mmx, -INFINITY, (long long)N * H);
      k_fill<<<g1d((long long)N * H), B, 0, stream>>>(den, 0.f, (long long)N * H);
      k_edge_e<<<g1d((long long)E2 * H), B, 0, stream>>>(asn, adn, src2, dst2, eE, mmx,
                                                         (long long)E2 * H, H);
      k_edge_ex<<<g1d((long long)E2 * H), B, 0, stream>>>(eE, mmx, den, dst2,
                                                          (long long)E2 * H, H);
      k_scatter<<<g1d((long long)E2 * H * C), B, 0, stream>>>(h2, eE, den, src2, dst2, xc,
                                                              (long long)E2 * H * C, H, C,
                                                              128, 640, 512);
    }
    // xc = elu(xc + [g1_b ; g2_b]); xa = elu(xa + xc @ uni_W + uni_b)
    k_xc_bias_elu<<<g1d((long long)N * 640), B, 0, stream>>>(xc, g1_b + (size_t)i * 512,
                                                             g2_b + (size_t)i * 128,
                                                             (long long)N * 640);
    gemm(xc, uniWt + (size_t)i * 640 * 512, h1, N, 512, 640);
    k_resid_elu<<<g1d((long long)N * 512), B, 0, stream>>>(xa, h1, uni_b + (size_t)i * 512,
                                                           (long long)N * 512, 512);
  }

  // ---- final LN + GAT (H=8, C=128, concat=False -> mean) ----
  k_layernorm<<<g1d((long long)N * 32), B, 0, stream>>>(xa, xn, fin_g, fin_b, N, 512);
  gemm(xn, fgWt, hblk, N, 1024, 512);
  {
    const int H = 8, C = 128;
    k_attn_node<<<g1d((long long)N * H), B, 0, stream>>>(hblk, fg_as, fg_ad, asn, adn,
                                                         N, H, C, 1024);
    k_fill<<<g1d((long long)N * H), B, 0, stream>>>(mmx, -INFINITY, (long long)N * H);
    k_fill<<<g1d((long long)N * H), B, 0, stream>>>(den, 0.f, (long long)N * H);
    k_edge_e<<<g1d((long long)E2 * H), B, 0, stream>>>(asn, adn, src2, dst2, eE, mmx,
                                                       (long long)E2 * H, H);
    k_edge_ex<<<g1d((long long)E2 * H), B, 0, stream>>>(eE, mmx, den, dst2,
                                                        (long long)E2 * H, H);
    k_out_bias<<<g1d((long long)N * C), B, 0, stream>>>(out, fg_b, (long long)N * C, C);
    k_scatter_mean<<<g1d((long long)E2 * C), B, 0, stream>>>(hblk, eE, den, src2, dst2, out,
                                                             (long long)E2 * C, H, C, 1024);
  }
}
